// Adamas_249108103397
// MI455X (gfx1250) — 8-bit matrix operand order repaired and checked by reading, compiled, not run on hardware
//
#include <hip/hip_runtime.h>
#include <math.h>

#define H     32
#define HKV   8
#define D     128
#define HIDDEN 4096
#define G     4
#define NT    32

typedef int   v8i __attribute__((ext_vector_type(8)));

__device__ __forceinline__ unsigned fkey(float f) {
    unsigned u = __float_as_uint(f);
    return (u & 0x80000000u) ? ~u : (u | 0x80000000u);
}

__device__ __forceinline__ void fwht128(float4& v, int lane) {
    float a0 = v.x + v.y, a1 = v.x - v.y, a2 = v.z + v.w, a3 = v.z - v.w;
    v.x = a0 + a2; v.y = a1 + a3; v.z = a0 - a2; v.w = a1 - a3;
#pragma unroll
    for (int mask = 1; mask <= 16; mask <<= 1) {
        float ox = __shfl_xor(v.x, mask, 32);
        float oy = __shfl_xor(v.y, mask, 32);
        float oz = __shfl_xor(v.z, mask, 32);
        float ow = __shfl_xor(v.w, mask, 32);
        if (lane & mask) { v.x = ox - v.x; v.y = oy - v.y; v.z = oz - v.z; v.w = ow - v.w; }
        else             { v.x += ox;      v.y += oy;      v.z += oz;      v.w += ow;      }
    }
}

__device__ __forceinline__ float waveSum(float s) {
#pragma unroll
    for (int m = 16; m; m >>= 1) s += __shfl_xor(s, m, 32);
    return s;
}

__device__ __forceinline__ int quantPack(float4 v, float s) {
    float inv = 1.0f / s;
    int c0 = (int)rintf(v.x * inv); c0 = max(-2, min(1, c0));
    int c1 = (int)rintf(v.y * inv); c1 = max(-2, min(1, c1));
    int c2 = (int)rintf(v.z * inv); c2 = max(-2, min(1, c2));
    int c3 = (int)rintf(v.w * inv); c3 = max(-2, min(1, c3));
    return (c0 & 255) | ((c1 & 255) << 8) | ((c2 & 255) << 16) | ((c3 & 255) << 24);
}

__global__ void gemv4096(const float* __restrict__ W, const float* __restrict__ x,
                         float* __restrict__ y, int rows) {
    __shared__ float xs[HIDDEN];
    for (int i = threadIdx.x; i < HIDDEN; i += blockDim.x) xs[i] = x[i];
    __syncthreads();
    int wave = threadIdx.x >> 5, lane = threadIdx.x & 31;
    int row = blockIdx.x * (blockDim.x >> 5) + wave;
    if (row >= rows) return;
    const float4* Wr = (const float4*)(W + (size_t)row * HIDDEN);
    const float4* Xr = (const float4*)xs;
    float acc = 0.f;
#pragma unroll 4
    for (int j = 0; j < 32; ++j) {
        float4 w = Wr[j * 32 + lane];
        float4 v = Xr[j * 32 + lane];
        acc += w.x * v.x + w.y * v.y + w.z * v.z + w.w * v.w;
    }
    acc = waveSum(acc);
    if (lane == 0) y[row] = acc;
}

__global__ void prep_kernel(const float* __restrict__ qkv,
                            float* __restrict__ q_rope,
                            float* __restrict__ k_new,
                            int*   __restrict__ cq8,
                            float* __restrict__ sq,
                            int S) {
    __shared__ float qs[H * D];
    __shared__ float ks[HKV * D];
    const float pos = (float)S;
    const float lnT = logf(10000.0f);

    for (int idx = threadIdx.x; idx < H * (D / 2); idx += blockDim.x) {
        int h = idx >> 6, i = idx & 63;
        float freq = expf(-(2.0f * (float)i / (float)D) * lnT);
        float c, s; __sincosf(pos * freq, &s, &c);
        float x1 = qkv[h * D + i], x2 = qkv[h * D + 64 + i];
        qs[h * D + i]      = x1 * c - x2 * s;
        qs[h * D + 64 + i] = x2 * c + x1 * s;
    }
    for (int idx = threadIdx.x; idx < HKV * (D / 2); idx += blockDim.x) {
        int h = idx >> 6, i = idx & 63;
        float freq = expf(-(2.0f * (float)i / (float)D) * lnT);
        float c, s; __sincosf(pos * freq, &s, &c);
        float x1 = qkv[H * D + h * D + i], x2 = qkv[H * D + h * D + 64 + i];
        ks[h * D + i]      = x1 * c - x2 * s;
        ks[h * D + 64 + i] = x2 * c + x1 * s;
    }
    __syncthreads();

    for (int i = threadIdx.x; i < H * D; i += blockDim.x)   q_rope[i] = qs[i];
    for (int i = threadIdx.x; i < HKV * D; i += blockDim.x) k_new[i]  = ks[i];

    int wave = threadIdx.x >> 5, lane = threadIdx.x & 31;
#pragma unroll
    for (int it = 0; it < 4; ++it) {
        int h = wave * 4 + it;
        float4 v = ((const float4*)(qs + h * D))[lane];
        fwht128(v, lane);
        float s = waveSum(fabsf(v.x) + fabsf(v.y) + fabsf(v.z) + fabsf(v.w));
        s = s * (1.0f / (float)D) + 1e-6f;
        cq8[h * 32 + lane] = quantPack(v, s);
        if (lane == 0) sq[h] = s;
    }
}

__global__ void score_kernel(const float* __restrict__ k_cache,
                             const int*   __restrict__ cq8,
                             const float* __restrict__ sq,
                             float* __restrict__ estG,
                             int S) {
    __shared__ int   ckL[HKV][NT][32];
    __shared__ float skL[HKV][NT];
    __shared__ int   cqL[H][32];
    __shared__ float sqL[H];

    for (int i = threadIdx.x; i < H * 32; i += blockDim.x) ((int*)cqL)[i] = cq8[i];
    for (int i = threadIdx.x; i < H;      i += blockDim.x) sqL[i] = sq[i];

    int wave = threadIdx.x >> 5, lane = threadIdx.x & 31;
    int tBase = blockIdx.x * NT;

    int n = wave;
    for (int tl = 0; tl < NT; ++tl) {
        int t = tBase + tl;
        float4 v = make_float4(0.f, 0.f, 0.f, 0.f);
        if (t < S) {
            const float4* kp = (const float4*)(k_cache + ((size_t)t * HKV + n) * D);
            v = kp[lane];
            if (t + 1 < S)
                __builtin_prefetch(k_cache + ((size_t)(t + 1) * HKV + n) * D + 4 * lane, 0, 1);
        }
        fwht128(v, lane);
        float s = waveSum(fabsf(v.x) + fabsf(v.y) + fabsf(v.z) + fabsf(v.w));
        s = s * (1.0f / (float)D) + 1e-6f;
        ckL[n][tl][lane] = (t < S) ? quantPack(v, s) : 0;
        if (lane == 0) skL[n][tl] = (t < S) ? s : 0.f;
    }
    __syncthreads();

    if (wave < 4) {
        int Mtile = wave >> 1;
        int Ntile = wave & 1;
        int hi = lane >> 4, mr = lane & 15;

        v8i acc = {0, 0, 0, 0, 0, 0, 0, 0};
#pragma unroll
        for (int s8 = 0; s8 < 8; ++s8) {
            int slice = Mtile * 8 + s8;
            int kn    = slice >> 1;
            int half  = slice & 1;

            int hA = Mtile * 16 + mr;
            bool valid = ((hA >> 2) == kn);
            v8i A;
#pragma unroll
            for (int r = 0; r < 8; ++r) {
                int kb = half * 64 + (r >> 2) * 32 + hi * 16 + (r & 3) * 4;
                A[r] = valid ? cqL[hA][kb >> 2] : 0;
            }
            int tl = Ntile * 16 + mr;
            v8i B;
#pragma unroll
            for (int r = 0; r < 8; ++r) {
                int kb = half * 64 + (r >> 2) * 32 + hi * 16 + (r & 3) * 4;
                B[r] = ckL[kn][tl][kb >> 2];
            }
            acc = __builtin_amdgcn_wmma_i32_16x16x64_iu8(
                       true, A,  true, B, acc,
                       false,  false);
        }

        int t = tBase + Ntile * 16 + mr;
        if (t < S) {
#pragma unroll
            for (int r = 0; r < 8; ++r) {
                int h   = Mtile * 16 + hi * 8 + r;
                int kvn = h >> 2;
                estG[(size_t)h * S + t] = (float)acc[r] * sqL[h] * skL[kvn][Ntile * 16 + mr];
            }
        }
    }
}

__global__ void topk_kernel(const float* __restrict__ estG, int* __restrict__ idxG,
                            int S, const int* __restrict__ topk_p) {
    int h = blockIdx.x;
    int topk = *topk_p;
    int need = topk - 1;
    const float* e = estG + (size_t)h * S;

    __shared__ unsigned hist[256];
    __shared__ int shBin, shAbove;

    unsigned prefix = 0;
    int target = need;
    for (int pass = 0; pass < 4; ++pass) {
        int shift = 24 - pass * 8;
        for (int i = threadIdx.x; i < 256; i += blockDim.x) hist[i] = 0u;
        __syncthreads();
        for (int i = threadIdx.x; i < S; i += blockDim.x) {
            unsigned key = fkey(e[i]);
            if (pass == 0 || (key >> (shift + 8)) == prefix)
                atomicAdd(&hist[(key >> shift) & 255u], 1u);
        }
        __syncthreads();
        if (threadIdx.x == 0) {
            unsigned cum = 0; int b = 255;
            for (; b > 0; --b) {
                unsigned c = hist[b];
                if (cum + c >= (unsigned)target) break;
                cum += c;
            }
            shBin = b; shAbove = (int)cum;
        }
        __syncthreads();
        prefix = (prefix << 8) | (unsigned)shBin;
        target -= shAbove;
        __syncthreads();
    }
    unsigned thresh = prefix;
    int gtCount = need - target;

    __shared__ int cGT, cEQ;
    if (threadIdx.x == 0) { cGT = 0; cEQ = 0; }
    __syncthreads();
    int* out = idxG + (size_t)h * topk;
    for (int i = threadIdx.x; i < S; i += blockDim.x) {
        unsigned key = fkey(e[i]);
        if (key > thresh) {
            int p = atomicAdd(&cGT, 1);
            out[1 + p] = i;
        } else if (key == thresh) {
            int p = atomicAdd(&cEQ, 1);
            if (p < target) out[1 + gtCount + p] = i;
        }
    }
    if (threadIdx.x == 0) out[0] = S;
}

__global__ void attn_kernel(const float* __restrict__ q_rope,
                            const float* __restrict__ k_cache,
                            const float* __restrict__ v_cache,
                            const float* __restrict__ k_new,
                            const float* __restrict__ v_new,
                            const int*   __restrict__ idxG,
                            float* __restrict__ attnOut,
                            int S, const int* __restrict__ topk_p) {
    int h = blockIdx.x;
    int n = h >> 2;
    int topk = *topk_p;
    int wave = threadIdx.x >> 5, lane = threadIdx.x & 31;
    const float scale = 0.08838834764831845f;

    float4 q = ((const float4*)(q_rope + h * D))[lane];

    float m = -__builtin_inff(), ssum = 0.f;
    float4 acc = make_float4(0.f, 0.f, 0.f, 0.f);
    const int* idx = idxG + (size_t)h * topk;

    for (int k = wave; k < topk; k += 8) {
        int id = idx[k];
        const float* Kp = (id < S) ? (k_cache + ((size_t)id * HKV + n) * D) : (k_new + n * D);
        const float* Vp = (id < S) ? (v_cache + ((size_t)id * HKV + n) * D) : (v_new + n * D);
        float4 kv = ((const float4*)Kp)[lane];
        float dot = q.x * kv.x + q.y * kv.y + q.z * kv.z + q.w * kv.w;
        dot = waveSum(dot);
        float logit = dot * scale;
        float4 vv = ((const float4*)Vp)[lane];
        float mn = fmaxf(m, logit);
        float sc = (ssum > 0.f) ? expf(m - mn) : 0.f;
        float p  = expf(logit - mn);
        ssum = ssum * sc + p;
        acc.x = acc.x * sc + p * vv.x;
        acc.y = acc.y * sc + p * vv.y;
        acc.z = acc.z * sc + p * vv.z;
        acc.w = acc.w * sc + p * vv.w;
        m = mn;
    }

    __shared__ float rm[8], rs[8];
    __shared__ float ra[8][D];
    ((float4*)ra[wave])[lane] = acc;
    if (lane == 0) { rm[wave] = m; rs[wave] = ssum; }
    __syncthreads();

    if (wave == 0) {
        float M = -__builtin_inff();
#pragma unroll
        for (int w = 0; w < 8; ++w) M = fmaxf(M, rm[w]);
        float tot = 0.f;
        float4 a = make_float4(0.f, 0.f, 0.f, 0.f);
#pragma unroll
        for (int w = 0; w < 8; ++w) {
            float sc = (rs[w] > 0.f) ? expf(rm[w] - M) : 0.f;
            tot += rs[w] * sc;
            float4 aw = ((const float4*)ra[w])[lane];
            a.x += aw.x * sc; a.y += aw.y * sc; a.z += aw.z * sc; a.w += aw.w * sc;
        }
        float inv = 1.0f / tot;
        float4 o = make_float4(a.x * inv, a.y * inv, a.z * inv, a.w * inv);
        ((float4*)(attnOut + h * D))[lane] = o;
    }
}

extern "C" void kernel_launch(void* const* d_in, const int* in_sizes, int n_in,
                              void* d_out, int out_size, void* d_ws, size_t ws_size,
                              hipStream_t stream) {
    const float* x       = (const float*)d_in[0];
    const float* Wq      = (const float*)d_in[1];
    const float* Wk      = (const float*)d_in[2];
    const float* Wv      = (const float*)d_in[3];
    const float* Wo      = (const float*)d_in[4];
    const float* k_cache = (const float*)d_in[5];
    const float* v_cache = (const float*)d_in[6];
    const int*   topk_p  = (const int*)d_in[7];
    float* out = (float*)d_out;

    const int S = in_sizes[5] / (HKV * D);

    float* W      = (float*)d_ws;
    float* qkv    = W;
    float* q_rope = W + 6144;
    float* k_new  = W + 10240;
    float* attn   = W + 11264;
    float* sq     = W + 15360;
    int*   cq8    = (int*)(W + 15392);
    float* est    = W + 16416;
    int*   idx    = (int*)(est + (size_t)H * S);
    const float* v_new = qkv + 5120;

    gemv4096<<<(HIDDEN + 7) / 8, 256, 0, stream>>>(Wq, x, qkv, HIDDEN);
    gemv4096<<<(HKV * D + 7) / 8, 256, 0, stream>>>(Wk, x, qkv + HIDDEN, HKV * D);
    gemv4096<<<(HKV * D + 7) / 8, 256, 0, stream>>>(Wv, x, qkv + HIDDEN + HKV * D, HKV * D);

    prep_kernel<<<1, 256, 0, stream>>>(qkv, q_rope, k_new, cq8, sq, S);

    score_kernel<<<(S + NT - 1) / NT, 256, 0, stream>>>(k_cache, cq8, sq, est, S);

    topk_kernel<<<H, 256, 0, stream>>>(est, idx, S, topk_p);

    attn_kernel<<<H, 256, 0, stream>>>(q_rope, k_cache, v_cache, k_new, v_new,
                                       idx, attn, S, topk_p);

    gemv4096<<<(HIDDEN + 7) / 8, 256, 0, stream>>>(Wo, attn, out, HIDDEN);
}
